// SchNetWrapper_28587302322447
// MI455X (gfx1250) — matrix operand order repaired and checked by reading, not compiled, not run on hardware
//
#include <hip/hip_runtime.h>
#include <hip/hip_bf16.h>

// ---------------------------------------------------------------------------
// SchNet forward on gfx1250 (MI455X). All GEMMs via v_wmma_f32_16x16x32_f16.
// Weights pre-packed into WMMA B-fragment image layout (32B/lane contiguous).
// N=8192 nodes, K=32 neighbors, HID=FIL=128, NG=50 (pad 64), LAYERS=6.
// ---------------------------------------------------------------------------

#define N_NODES   8192
#define ATOMS     128
#define NGRAPH    64
#define KNBR      32
#define HID       128
#define NGPAD     64
#define LAYERS    6
#define CUT       10.0f
#define PI_F      3.14159265f

typedef __attribute__((ext_vector_type(16))) _Float16 v16h;
typedef __attribute__((ext_vector_type(8)))  float    v8f;

__device__ __forceinline__ float silu_f(float x) {
    return x / (1.0f + expf(-x));
}

// A-matrix fragment (16x32 f16) from row-major LDS tile, leading dim ld.
// ISA 7.12.2: lanes 0-15 row M=lane, elems 0-7 -> K 0-7, 8-15 -> K 16-23;
// lanes 16-31 same rows, K shifted +8. Lane data is two contiguous 16B runs
// -> compiler emits ds_load_b128 pairs.
__device__ __forceinline__ v16h load_a_frag(const _Float16* base, int ld) {
    int lane  = threadIdx.x & 31;
    int row   = lane & 15;
    int half8 = (lane >> 4) << 3;           // 0 or 8
    v16h a;
#pragma unroll
    for (int e = 0; e < 16; ++e) {
        int k = (e & 7) + ((e >> 3) << 4) + half8;
        a[e] = base[row * ld + k];
    }
    return a;
}

// B fragment from packed fragment-image weights: tile (n0, kk) of a KxN
// weight with ktiles = K/32. Each tile stores 32 lanes x 16 halfs
// contiguously -> one 32B aligned v16h load per lane (2x global_load_b128).
__device__ __forceinline__ v16h load_b_packed(const _Float16* wp, int n0, int ktiles, int kk) {
    int lane = threadIdx.x & 31;
    const v16h* p = (const v16h*)(wp + (size_t)((((n0 >> 4) * ktiles) + kk) << 9) + lane * 16);
    return *p;
}

__device__ __forceinline__ v8f wmma_f16f32(v16h a, v16h b, v8f c) {
    return __builtin_amdgcn_wmma_f32_16x16x32_f16(
        /*neg_a=*/false, a, /*neg_b=*/false, b,
        /*c_mod=*/(short)0, c, /*reuse_a=*/false, /*reuse_b=*/false);
}

// Inverse of the packed layout: flat index r within one layer's packed weight
// (K x 128) -> (k, n) in the original row-major weight.
__device__ __forceinline__ void unpack_idx(int r, int K, int* k, int* n) {
    int tile = r >> 9;          // /512 halfs per 32x16 tile
    int rr   = r & 511;
    int lane = rr >> 4;
    int e    = rr & 15;
    int ktiles = K >> 5;
    int n0 = (tile / ktiles) << 4;
    int k0 = (tile % ktiles) << 5;
    *k = k0 + ((lane >> 4) << 3) + (e & 7) + ((e >> 3) << 4);
    *n = n0 + (lane & 15);
}

// ---------------------------------------------------------------------------
// Weight conversion fp32 -> packed fp16 fragment images.
// mw1 padded NG 50 -> 64 (zeros).
// ---------------------------------------------------------------------------
__global__ void conv_weights_k(const float* __restrict__ mw1,
                               const float* __restrict__ mw2,
                               const float* __restrict__ l1w,
                               const float* __restrict__ l2w,
                               _Float16* __restrict__ mw1p,
                               _Float16* __restrict__ mw2p,
                               _Float16* __restrict__ l1wp,
                               _Float16* __restrict__ l2wp) {
    int t = blockIdx.x * blockDim.x + threadIdx.x;        // 0 .. 98303
    if (t < LAYERS * NGPAD * HID) {                       // 49152
        int l = t / (NGPAD * HID);
        int r = t % (NGPAD * HID);
        int k, n; unpack_idx(r, NGPAD, &k, &n);
        mw1p[t] = (_Float16)((k < 50) ? mw1[(l * 50 + k) * HID + n] : 0.0f);
    }
    if (t < LAYERS * HID * HID) {                         // 98304
        int l = t / (HID * HID);
        int r = t % (HID * HID);
        int k, n; unpack_idx(r, HID, &k, &n);
        size_t s = (size_t)(l * HID + k) * HID + n;
        mw2p[t] = (_Float16)mw2[s];
        l1wp[t] = (_Float16)l1w[s];
        l2wp[t] = (_Float16)l2w[s];
    }
}

// ---------------------------------------------------------------------------
// Graph build: one block per graph (128 threads = 128 centers). Selection of
// the 32 nearest same-graph neighbors within cutoff (lowest-index tie-break,
// matching top_k). Invalid slots: idx=self, cval=0 -> contribute nothing.
// ---------------------------------------------------------------------------
__global__ void build_graph_k(const float* __restrict__ pos,
                              int* __restrict__ idx,
                              float* __restrict__ dist,
                              float* __restrict__ cval) {
    __shared__ float px[ATOMS], py[ATOMS], pz[ATOMS];
    int g = blockIdx.x;
    int i = threadIdx.x;
    int node = g * ATOMS + i;
    px[i] = pos[node * 3 + 0];
    py[i] = pos[node * 3 + 1];
    pz[i] = pos[node * 3 + 2];
    __syncthreads();

    unsigned ch0 = 0, ch1 = 0, ch2 = 0, ch3 = 0;
    {   // mark self as taken
        unsigned bit = 1u << (i & 31);
        if (i < 32) ch0 |= bit; else if (i < 64) ch1 |= bit;
        else if (i < 96) ch2 |= bit; else ch3 |= bit;
    }
    float xi = px[i], yi = py[i], zi = pz[i];
    bool dead = false;

    for (int k = 0; k < KNBR; ++k) {
        float bd = 3.0e38f;
        int   bj = -1;
        if (!dead) {
            for (int j = 0; j < ATOMS; ++j) {
                unsigned word = (j < 32) ? ch0 : (j < 64) ? ch1 : (j < 96) ? ch2 : ch3;
                if ((word >> (j & 31)) & 1u) continue;
                float dx = px[j] - xi, dy = py[j] - yi, dz = pz[j] - zi;
                float d2 = dx * dx + dy * dy + dz * dz;
                if (d2 < bd) { bd = d2; bj = j; }
            }
        }
        int o = node * KNBR + k;
        if (bj >= 0 && bd < CUT * CUT) {
            unsigned bit = 1u << (bj & 31);
            if (bj < 32) ch0 |= bit; else if (bj < 64) ch1 |= bit;
            else if (bj < 96) ch2 |= bit; else ch3 |= bit;
            float d = sqrtf(bd);
            idx[o]  = g * ATOMS + bj;
            dist[o] = d;
            cval[o] = 0.5f * (cosf(d * PI_F / CUT) + 1.0f);
        } else {
            dead = true;
            idx[o]  = node;     // self; message zeroed by cval=0
            dist[o] = 1.0f;
            cval[o] = 0.0f;
        }
    }
}

// ---------------------------------------------------------------------------
// Embedding gather: h = emb[z] (fp32 + fp16 shadow).
// ---------------------------------------------------------------------------
__global__ void embed_k(const int* __restrict__ z,
                        const float* __restrict__ emb,
                        float* __restrict__ h,
                        _Float16* __restrict__ h16) {
    int t = blockIdx.x * blockDim.x + threadIdx.x;   // N_NODES*HID
    int n = t >> 7;
    int f = t & 127;
    float v = emb[z[n] * HID + f];
    h[t]   = v;
    h16[t] = (_Float16)v;
}

// ---------------------------------------------------------------------------
// Node linear 1: xj = h16 @ l1w[l] + l1b[l].  One block = 32 rows x 128 cols;
// 8 waves each own a 32x16 strip (two M-tiles sharing each B fragment).
// ---------------------------------------------------------------------------
__global__ void node_lin1_k(const _Float16* __restrict__ h16,
                            const _Float16* __restrict__ wp,
                            const float* __restrict__ b,
                            float* __restrict__ xj) {
    __shared__ __align__(16) _Float16 As[32 * HID];
    int row0 = blockIdx.x * 32;
    int t = threadIdx.x;
    ((uint4*)As)[t]       = ((const uint4*)(h16 + (size_t)row0 * HID))[t];
    ((uint4*)As)[t + 256] = ((const uint4*)(h16 + (size_t)row0 * HID))[t + 256];
    __syncthreads();

    int lane = t & 31;
    int n0   = (t >> 5) * 16;
    int col  = n0 + (lane & 15);
    int madd = (lane >> 4) << 3;

    v8f c0, c1;
    float bias = b[col];
#pragma unroll
    for (int r = 0; r < 8; ++r) { c0[r] = bias; c1[r] = bias; }
#pragma unroll
    for (int kk = 0; kk < 4; ++kk) {
        v16h bb = load_b_packed(wp, n0, 4, kk);
        v16h a0 = load_a_frag(As + kk * 32, HID);
        v16h a1 = load_a_frag(As + 16 * HID + kk * 32, HID);
        c0 = wmma_f16f32(a0, bb, c0);
        c1 = wmma_f16f32(a1, bb, c1);
    }
#pragma unroll
    for (int r = 0; r < 8; ++r) {
        int m = r + madd;
        xj[(size_t)(row0 + m) * HID + col]      = c0[r];
        xj[(size_t)(row0 + 16 + m) * HID + col] = c1[r];
    }
}

// ---------------------------------------------------------------------------
// Node linear 2 + residual: h += silu(agg @ l2w[l] + l2b[l]); refresh h16.
// ---------------------------------------------------------------------------
__global__ void node_lin2_k(const float* __restrict__ agg,
                            const _Float16* __restrict__ wp,
                            const float* __restrict__ b,
                            float* __restrict__ h,
                            _Float16* __restrict__ h16) {
    __shared__ __align__(16) _Float16 As[32 * HID];
    int row0 = blockIdx.x * 32;
    int t = threadIdx.x;
#pragma unroll
    for (int q = 0; q < 16; ++q) {
        int s = t + 256 * q;
        As[s] = (_Float16)agg[(size_t)row0 * HID + s];
    }
    __syncthreads();

    int lane = t & 31;
    int n0   = (t >> 5) * 16;
    int col  = n0 + (lane & 15);
    int madd = (lane >> 4) << 3;

    v8f c0, c1;
    float bias = b[col];
#pragma unroll
    for (int r = 0; r < 8; ++r) { c0[r] = bias; c1[r] = bias; }
#pragma unroll
    for (int kk = 0; kk < 4; ++kk) {
        v16h bb = load_b_packed(wp, n0, 4, kk);
        v16h a0 = load_a_frag(As + kk * 32, HID);
        v16h a1 = load_a_frag(As + 16 * HID + kk * 32, HID);
        c0 = wmma_f16f32(a0, bb, c0);
        c1 = wmma_f16f32(a1, bb, c1);
    }
#pragma unroll
    for (int r = 0; r < 8; ++r) {
        int m = r + madd;
        size_t p0 = (size_t)(row0 + m) * HID + col;
        size_t p1 = (size_t)(row0 + 16 + m) * HID + col;
        float h0 = h[p0] + silu_f(c0[r]);
        float h1 = h[p1] + silu_f(c1[r]);
        h[p0] = h0;  h16[p0] = (_Float16)h0;
        h[p1] = h1;  h16[p1] = (_Float16)h1;
    }
}

// ---------------------------------------------------------------------------
// Edge kernel: one block = all 32 edges of one sender i (= blockIdx.x).
//   A1   = rbf(dist)          [32 x 64]  f16 (gaussians 50, zero-padded)
//   T1   = silu(A1@mw1 + b1)  [32 x 128] f16
//   W    = T1@mw2 + b2        [32 x 128] f32 (in regs)
//   msg  = W * cval[e] * xj[i][:]  -> atomic add into agg[recv[e]][:]
// 8 waves; wave w owns cols [16w,16w+16), both 16-row tiles (12 wmma/wave).
// ---------------------------------------------------------------------------
__global__ void edge_k(const int* __restrict__ idx,
                       const float* __restrict__ dist,
                       const float* __restrict__ cval,
                       const _Float16* __restrict__ mw1p,
                       const float* __restrict__ mb1,
                       const _Float16* __restrict__ mw2p,
                       const float* __restrict__ mb2,
                       const float* __restrict__ xj,
                       float* __restrict__ agg) {
    __shared__ __align__(16) _Float16 A1[32 * NGPAD];
    __shared__ __align__(16) _Float16 T1[32 * HID];
    __shared__ float dist_s[32], cval_s[32];
    __shared__ int   recv_s[32];
    __shared__ float xj_s[HID];

    int snd = blockIdx.x;            // sender node; edges snd*32 .. +31
    int e0  = snd * KNBR;
    int t   = threadIdx.x;

    if (t < 32) {
        dist_s[t] = dist[e0 + t];
        cval_s[t] = cval[e0 + t];
        recv_s[t] = idx[e0 + t];
    }
    if (t < HID) xj_s[t] = xj[(size_t)snd * HID + t];
    __syncthreads();

    // RBF expansion: linspace(0,10,50), width = 10/49
    const float width = CUT / 49.0f;
#pragma unroll
    for (int q = 0; q < 8; ++q) {
        int s = t + 256 * q;         // 2048 = 32*64
        int m = s >> 6;
        int g = s & 63;
        float v = 0.0f;
        if (g < 50) {
            float off = width * (float)g;
            float u = (dist_s[m] - off) / width;
            v = expf(-0.5f * u * u);
        }
        A1[s] = (_Float16)v;
    }
    __syncthreads();

    int lane = t & 31;
    int n0   = (t >> 5) * 16;
    int col  = n0 + (lane & 15);
    int madd = (lane >> 4) << 3;

    // GEMM1: [32x64] @ [64x128] + mb1
    v8f c0, c1;
    float b1 = mb1[col];
#pragma unroll
    for (int r = 0; r < 8; ++r) { c0[r] = b1; c1[r] = b1; }
#pragma unroll
    for (int kk = 0; kk < 2; ++kk) {
        v16h bb = load_b_packed(mw1p, n0, 2, kk);
        v16h a0 = load_a_frag(A1 + kk * 32, NGPAD);
        v16h a1 = load_a_frag(A1 + 16 * NGPAD + kk * 32, NGPAD);
        c0 = wmma_f16f32(a0, bb, c0);
        c1 = wmma_f16f32(a1, bb, c1);
    }
    // SiLU -> T1
#pragma unroll
    for (int r = 0; r < 8; ++r) {
        int m = r + madd;
        T1[m * HID + col]        = (_Float16)silu_f(c0[r]);
        T1[(m + 16) * HID + col] = (_Float16)silu_f(c1[r]);
    }
    __syncthreads();

    // GEMM2: [32x128] @ [128x128] + mb2
    float b2 = mb2[col];
#pragma unroll
    for (int r = 0; r < 8; ++r) { c0[r] = b2; c1[r] = b2; }
#pragma unroll
    for (int kk = 0; kk < 4; ++kk) {
        v16h bb = load_b_packed(mw2p, n0, 4, kk);
        v16h a0 = load_a_frag(T1 + kk * 32, HID);
        v16h a1 = load_a_frag(T1 + 16 * HID + kk * 32, HID);
        c0 = wmma_f16f32(a0, bb, c0);
        c1 = wmma_f16f32(a1, bb, c1);
    }

    // msg = W * cval * xj[sender], scatter-add at receiver
#pragma unroll
    for (int r = 0; r < 8; ++r) {
        int m = r + madd;
        float v0 = c0[r] * cval_s[m] * xj_s[col];
        float v1 = c1[r] * cval_s[m + 16] * xj_s[col];
        atomicAdd(&agg[(size_t)recv_s[m] * HID + col], v0);
        atomicAdd(&agg[(size_t)recv_s[m + 16] * HID + col], v1);
    }
}

// ---------------------------------------------------------------------------
// Output MLP + per-graph segment sum. One block (64 threads) per node.
// ---------------------------------------------------------------------------
__global__ void out_k(const float* __restrict__ h,
                      const int* __restrict__ batch,
                      const float* __restrict__ ow1,
                      const float* __restrict__ ob1,
                      const float* __restrict__ ow2,
                      const float* __restrict__ ob2,
                      float* __restrict__ out) {
    __shared__ float hrow[HID];
    __shared__ float red[64];
    int n = blockIdx.x;
    int t = threadIdx.x;
    hrow[t]      = h[(size_t)n * HID + t];
    hrow[t + 64] = h[(size_t)n * HID + 64 + t];
    __syncthreads();

    float acc = ob1[t];
#pragma unroll 8
    for (int j = 0; j < HID; ++j)
        acc += hrow[j] * ow1[j * 64 + t];
    red[t] = silu_f(acc) * ow2[t];
    __syncthreads();
    for (int s = 32; s > 0; s >>= 1) {
        if (t < s) red[t] += red[t + s];
        __syncthreads();
    }
    if (t == 0) atomicAdd(&out[batch[n]], red[0] + ob2[0]);
}

// ---------------------------------------------------------------------------
extern "C" void kernel_launch(void* const* d_in, const int* in_sizes, int n_in,
                              void* d_out, int out_size, void* d_ws, size_t ws_size,
                              hipStream_t stream) {
    (void)in_sizes; (void)n_in; (void)out_size; (void)ws_size;

    const float* pos  = (const float*)d_in[0];
    const int*   z    = (const int*)  d_in[1];
    const int*   bat  = (const int*)  d_in[2];
    const float* emb  = (const float*)d_in[3];
    const float* mw1  = (const float*)d_in[4];
    const float* mb1  = (const float*)d_in[5];
    const float* mw2  = (const float*)d_in[6];
    const float* mb2  = (const float*)d_in[7];
    const float* l1w  = (const float*)d_in[8];
    const float* l1b  = (const float*)d_in[9];
    const float* l2w  = (const float*)d_in[10];
    const float* l2b  = (const float*)d_in[11];
    const float* ow1  = (const float*)d_in[12];
    const float* ob1  = (const float*)d_in[13];
    const float* ow2  = (const float*)d_in[14];
    const float* ob2  = (const float*)d_in[15];
    float* out = (float*)d_out;

    // Workspace bump allocator (256 B aligned partitions).
    char* ws = (char*)d_ws;
    size_t off = 0;
    auto alloc = [&](size_t bytes) -> char* {
        char* p = ws + off;
        off = (off + bytes + 255) & ~(size_t)255;
        return p;
    };
    _Float16* mw1p = (_Float16*)alloc((size_t)LAYERS * NGPAD * HID * 2);
    _Float16* mw2p = (_Float16*)alloc((size_t)LAYERS * HID * HID * 2);
    _Float16* l1wp = (_Float16*)alloc((size_t)LAYERS * HID * HID * 2);
    _Float16* l2wp = (_Float16*)alloc((size_t)LAYERS * HID * HID * 2);
    float*    h    = (float*)   alloc((size_t)N_NODES * HID * 4);
    _Float16* h16  = (_Float16*)alloc((size_t)N_NODES * HID * 2);
    float*    xj   = (float*)   alloc((size_t)N_NODES * HID * 4);
    float*    agg  = (float*)   alloc((size_t)N_NODES * HID * 4);
    int*      idxb = (int*)     alloc((size_t)N_NODES * KNBR * 4);
    float*    dstb = (float*)   alloc((size_t)N_NODES * KNBR * 4);
    float*    cvlb = (float*)   alloc((size_t)N_NODES * KNBR * 4);

    conv_weights_k<<<384, 256, 0, stream>>>(mw1, mw2, l1w, l2w, mw1p, mw2p, l1wp, l2wp);
    build_graph_k<<<NGRAPH, ATOMS, 0, stream>>>(pos, idxb, dstb, cvlb);
    embed_k<<<(N_NODES * HID) / 256, 256, 0, stream>>>(z, emb, h, h16);
    hipMemsetAsync(out, 0, NGRAPH * sizeof(float), stream);

    const int nodeBlocks = N_NODES / 32;     // 256
    const int edgeBlocks = N_NODES;          // one sender per block
    for (int l = 0; l < LAYERS; ++l) {
        node_lin1_k<<<nodeBlocks, 256, 0, stream>>>(
            h16, l1wp + (size_t)l * HID * HID, l1b + l * HID, xj);
        hipMemsetAsync(agg, 0, (size_t)N_NODES * HID * sizeof(float), stream);
        edge_k<<<edgeBlocks, 256, 0, stream>>>(
            idxb, dstb, cvlb,
            mw1p + (size_t)l * NGPAD * HID, mb1 + l * HID,
            mw2p + (size_t)l * HID * HID,   mb2 + l * HID,
            xj, agg);
        node_lin2_k<<<nodeBlocks, 256, 0, stream>>>(
            agg, l2wp + (size_t)l * HID * HID, l2b + l * HID, h, h16);
    }
    out_k<<<N_NODES, 64, 0, stream>>>(h, bat, ow1, ob1, ow2, ob2, out);
}
